// EEGLSTMClassifier_80659485819496
// MI455X (gfx1250) — matrix operand order repaired and checked by reading, not compiled, not run on hardware
//
#include <hip/hip_runtime.h>
#include <hip/hip_bf16.h>
#include <math.h>

typedef _Float16 f16;
typedef __attribute__((ext_vector_type(16))) _Float16 v16h;
typedef __attribute__((ext_vector_type(8)))  float    v8f;

// ---- fragment load helpers -------------------------------------------------
// A fragment (16x32 MxK tile, this lane's 16 values): 8 contiguous f32 at
// k0+kOff and 8 more at +16, converted f32->f16.
__device__ __forceinline__ v16h load_a_frag(const float* __restrict__ Arow,
                                            int k0, int kOff)
{
    v16h a;
    const float2* p = (const float2*)(Arow + k0 + kOff);
    float2 q;
    q = p[0];  a[0]  = (f16)q.x; a[1]  = (f16)q.y;
    q = p[1];  a[2]  = (f16)q.x; a[3]  = (f16)q.y;
    q = p[2];  a[4]  = (f16)q.x; a[5]  = (f16)q.y;
    q = p[3];  a[6]  = (f16)q.x; a[7]  = (f16)q.y;
    q = p[8];  a[8]  = (f16)q.x; a[9]  = (f16)q.y;
    q = p[9];  a[10] = (f16)q.x; a[11] = (f16)q.y;
    q = p[10]; a[12] = (f16)q.x; a[13] = (f16)q.y;
    q = p[11]; a[14] = (f16)q.x; a[15] = (f16)q.y;
    return a;
}

// B fragment (32x16 KxN tiles x2): column n = n0 + lane%16 (+16), 8 k at kB and 8 more at kB+16.
__device__ __forceinline__ void load_b_frag_vec(const float* __restrict__ Wt,
                                                long long ldw, int kB, int n0,
                                                v16h& b0, v16h& b1)
{
    const float* pw = Wt + (long long)kB * ldw + n0 + (threadIdx.x & 15);
#pragma unroll
    for (int s = 0; s < 8; ++s) {
        b0[s]     = (f16)pw[(long long)s * ldw];
        b1[s]     = (f16)pw[(long long)s * ldw + 16];
        b0[s + 8] = (f16)pw[(long long)(s + 16) * ldw];
        b1[s + 8] = (f16)pw[(long long)(s + 16) * ldw + 16];
    }
}

// ---------------------------------------------------------------------------
// Generic batched WMMA GEMM:  C = act(alpha * A(MxK) * Wt + bias1 + bias2)
//   A  : f32 row-major (lda).   Wt : f32 K-major, element (k,n) at Wt[k*ldw+n].
//   blockIdx.z -> (b = z/nH, h = z%nH) pointer offsets.
// Each wave computes a 16x32 C strip (two 16x16 WMMA tiles, f32 accumulate).
// Branch-free K-loop: OOB rows/cols are address-clamped (their C entries are
// never stored); K-tail uses value-selects.  EXEC stays all-1s at every WMMA.
// nFull is wave-uniform -> scalarized via readfirstlane + loop unswitching, so
// the steady-state loop has no EXEC manipulation at all.
// ---------------------------------------------------------------------------
__global__ void __launch_bounds__(256)
gemm_wmma(const float* __restrict__ A, long long lda, long long sAb, long long sAh,
          const float* __restrict__ Wt, long long ldw, long long sWb, long long sWh,
          const float* __restrict__ bias1, const float* __restrict__ bias2,
          float* __restrict__ C, long long ldc, long long sCb, long long sCh,
          int M, int N, int K, float alpha, int act, int nH)
{
    const int z  = blockIdx.z;
    const int bb = z / nH;
    const int hh = z % nH;
    A  += (long long)bb * sAb + (long long)hh * sAh;
    Wt += (long long)bb * sWb + (long long)hh * sWh;
    C  += (long long)bb * sCb + (long long)hh * sCh;

    const int lane    = threadIdx.x & 31;
    const int wave    = threadIdx.x >> 5;            // 8 waves: 4 (M) x 2 (N)
    const int m0      = blockIdx.y * 64 + (wave >> 1) * 16;
    const int n0      = blockIdx.x * 64 + (wave & 1) * 32;
    const bool laneHi = (lane & 16) != 0;
    const int  laneLo = lane & 15;

    // Address-clamped A row (garbage rows only feed unstored C entries).
    const int mA = min(m0 + laneLo, M - 1);
    const float* __restrict__ Arow = A + (long long)mA * lda;
    const int kOffA = laneHi ? 8 : 0;

    v8f acc0 = {}, acc1 = {};
    const int Kmain = K & ~31;
    // wave-uniform -> force into an SGPR so the loops branch scalar-side
    const int nFull = __builtin_amdgcn_readfirstlane((n0 + 32 <= N) ? 1 : 0);

    if (nFull) {
        for (int k0 = 0; k0 < Kmain; k0 += 32) {
            __builtin_prefetch(Arow + k0 + 64, 0, 3);
            v16h b0, b1;
            v16h a = load_a_frag(Arow, k0, kOffA);
            load_b_frag_vec(Wt, ldw, k0 + kOffA, n0, b0, b1);
            acc0 = __builtin_amdgcn_wmma_f32_16x16x32_f16(false, a, false, b0, (short)0, acc0, false, false);
            acc1 = __builtin_amdgcn_wmma_f32_16x16x32_f16(false, a, false, b1, (short)0, acc1, false, false);
        }
    } else {
        for (int k0 = 0; k0 < Kmain; k0 += 32) {
            v16h b0, b1;
            v16h a = load_a_frag(Arow, k0, kOffA);
            const float* wr = Wt + (long long)(k0 + kOffA) * ldw;
            const int nc0 = min(n0 + laneLo, N - 1), nc1 = min(n0 + 16 + laneLo, N - 1);
#pragma unroll
            for (int j = 0; j < 16; ++j) {
                const long long kk = (j < 8) ? j : j + 8;
                b0[j] = (f16)wr[kk * ldw + nc0];
                b1[j] = (f16)wr[kk * ldw + nc1];
            }
            acc0 = __builtin_amdgcn_wmma_f32_16x16x32_f16(false, a, false, b0, (short)0, acc0, false, false);
            acc1 = __builtin_amdgcn_wmma_f32_16x16x32_f16(false, a, false, b1, (short)0, acc1, false, false);
        }
    }

    // K tail (K % 32): value-selects, still branch-free.
    if (Kmain < K) {
        v16h a, b0, b1;
        const int kb = Kmain + kOffA;
#pragma unroll
        for (int s = 0; s < 8; ++s) {
            const int k1 = kb + s, k2 = kb + 16 + s;
            const float v1 = Arow[min(k1, K - 1)];
            const float v2 = Arow[min(k2, K - 1)];
            a[s]     = (k1 < K) ? (f16)v1 : (f16)0.0f;
            a[s + 8] = (k2 < K) ? (f16)v2 : (f16)0.0f;
        }
        const int nc0 = min(n0 + laneLo, N - 1), nc1 = min(n0 + 16 + laneLo, N - 1);
#pragma unroll
        for (int j = 0; j < 16; ++j) {
            const int  kB  = kb + ((j < 8) ? j : j + 8);
            const bool okK = kB < K;
            const float* wr = Wt + (long long)min(kB, K - 1) * ldw;
            const float w0 = wr[nc0];
            const float w1 = wr[nc1];
            b0[j] = okK ? (f16)w0 : (f16)0.0f;
            b1[j] = okK ? (f16)w1 : (f16)0.0f;
        }
        acc0 = __builtin_amdgcn_wmma_f32_16x16x32_f16(false, a, false, b0, (short)0, acc0, false, false);
        acc1 = __builtin_amdgcn_wmma_f32_16x16x32_f16(false, a, false, b1, (short)0, acc1, false, false);
    }

    // Store: VGPR r -> M = m0 + r (+8 for lanes 16-31), N = n0 + (lane&15) (+16)
    const int nC0 = n0 + laneLo;
    const int nC1 = n0 + 16 + laneLo;
    float bv0 = 0.0f, bv1 = 0.0f;
    if (bias1) { if (nC0 < N) bv0 += bias1[nC0]; if (nC1 < N) bv1 += bias1[nC1]; }
    if (bias2) { if (nC0 < N) bv0 += bias2[nC0]; if (nC1 < N) bv1 += bias2[nC1]; }
    const int mBase = m0 + (laneHi ? 8 : 0);
#pragma unroll
    for (int r = 0; r < 8; ++r) {
        const int m = mBase + r;
        if (m < M) {
            if (nC0 < N) {
                float v = alpha * acc0[r] + bv0;
                if (act) v = fmaxf(v, 0.0f);
                C[(long long)m * ldc + nC0] = v;
            }
            if (nC1 < N) {
                float v = alpha * acc1[r] + bv1;
                if (act) v = fmaxf(v, 0.0f);
                C[(long long)m * ldc + nC1] = v;
            }
        }
    }
}

// (N x K) row-major -> (K x N) with destination leading dim dstLd
__global__ void transpose_nk(const float* __restrict__ src, float* __restrict__ dst,
                             int N, int K, int dstLd)
{
    int i = blockIdx.x * 256 + threadIdx.x;
    if (i >= N * K) return;
    int k = i % K, n = i / K;
    dst[(long long)k * dstLd + n] = src[i];
}

// (B,L,D) -> (B,D,L)
__global__ void transpose_ld(const float* __restrict__ src, float* __restrict__ dst,
                             int B, int L, int D)
{
    long long i = (long long)blockIdx.x * 256 + threadIdx.x;
    long long total = (long long)B * L * D;
    if (i >= total) return;
    int d = (int)(i % D);
    long long l = (i / D) % L;
    int b = (int)(i / ((long long)D * L));
    dst[((long long)b * D + d) * L + l] = src[i];
}

// ---------------------------------------------------------------------------
// Conv1d(k=3, SAME) + BatchNorm(eval, mean=0,var=1) + ReLU.  x:(B,Cin,L) y:(B,Cout,L)
// ---------------------------------------------------------------------------
__global__ void conv_bn_relu(const float* __restrict__ x, const float* __restrict__ w,
                             const float* __restrict__ cb,
                             const float* __restrict__ bng, const float* __restrict__ bnb,
                             float* __restrict__ y, int B, int Cin, int Cout, int L)
{
    long long i = (long long)blockIdx.x * blockDim.x + threadIdx.x;
    long long total = (long long)B * Cout * L;
    if (i >= total) return;
    int l = (int)(i % L);
    int o = (int)((i / L) % Cout);
    int b = (int)(i / ((long long)L * Cout));
    float acc = cb[o];
    const float* wr = w + (long long)o * Cin * 3;
    const float* xb = x + (long long)b * Cin * L;
    for (int ci = 0; ci < Cin; ++ci) {
        const float* xr = xb + (long long)ci * L;
        float x0 = (l > 0)     ? xr[l - 1] : 0.0f;
        float x1 = xr[l];
        float x2 = (l < L - 1) ? xr[l + 1] : 0.0f;
        acc += x0 * wr[ci * 3 + 0] + x1 * wr[ci * 3 + 1] + x2 * wr[ci * 3 + 2];
    }
    float inv = rsqrtf(1.0f + 1e-5f);
    float v = bng[o] * (acc * inv) + bnb[o];
    y[i] = v > 0.0f ? v : 0.0f;
}

// (B,C,L) -> (B,L,C)
__global__ void nchw_to_nlc(const float* __restrict__ src, float* __restrict__ dst,
                            int B, int C, int L)
{
    long long i = (long long)blockIdx.x * blockDim.x + threadIdx.x;
    long long total = (long long)B * C * L;
    if (i >= total) return;
    int c = (int)(i % C);
    long long l = (i / C) % L;
    int b = (int)(i / ((long long)C * L));
    dst[i] = src[((long long)b * C + c) * L + l];
}

// ---------------------------------------------------------------------------
// Persistent LSTM recurrence. blockIdx.x = direction (0 fwd, 1 bwd).
// pre*: (B,L,512) precomputed x@Wih^T + bih + bhh.  Whh: (512,128) (L2-resident).
// out: (B,L,256); dir writes columns [dir*128, dir*128+128).  H=128, B=2.
// ---------------------------------------------------------------------------
__device__ __forceinline__ float sigmoidf_(float x) { return 1.0f / (1.0f + __expf(-x)); }

__global__ void __launch_bounds__(256)
lstm_rec(const float* __restrict__ preF, const float* __restrict__ preB,
         const float* __restrict__ WhhF, const float* __restrict__ WhhB,
         float* __restrict__ out, int Lc)
{
    const int dir = blockIdx.x;
    const float* pre = dir ? preB : preF;
    const float* Whh = dir ? WhhB : WhhF;

    __shared__ float hS[2][128];
    __shared__ float cS[2][128];
    __shared__ float gS[2][512];

    const int tid = threadIdx.x;
    {
        int b = tid >> 7, u = tid & 127;
        hS[b][u] = 0.0f;
        cS[b][u] = 0.0f;
    }
    __syncthreads();

    for (int s = 0; s < Lc; ++s) {
        const int t = dir ? (Lc - 1 - s) : s;
#pragma unroll
        for (int q = 0; q < 4; ++q) {
            int idx = tid + q * 256;
            int b = idx >> 9;
            int j = idx & 511;
            const float* wr = Whh + (long long)j * 128;
            const float* hb = hS[b];
            float acc = pre[((long long)b * Lc + t) * 512 + j];
#pragma unroll 4
            for (int k = 0; k < 128; ++k) acc += hb[k] * wr[k];
            gS[b][j] = acc;
        }
        __syncthreads();
        {
            int b = tid >> 7, u = tid & 127;
            float ig = sigmoidf_(gS[b][u]);
            float fg = sigmoidf_(gS[b][128 + u]);
            float gg = tanhf(gS[b][256 + u]);
            float og = sigmoidf_(gS[b][384 + u]);
            float cc = fg * cS[b][u] + ig * gg;
            cS[b][u] = cc;
            float hh = og * tanhf(cc);
            hS[b][u] = hh;
            out[((long long)b * Lc + t) * 256 + dir * 128 + u] = hh;
        }
        __syncthreads();
    }
}

// Row softmax in place (one block per row)
__global__ void __launch_bounds__(256)
softmax_rows(float* __restrict__ p0, int n)
{
    __shared__ float red[256];
    float* p = p0 + (long long)blockIdx.x * n;
    int tid = threadIdx.x;
    float mx = -3.0e38f;
    for (int i = tid; i < n; i += 256) mx = fmaxf(mx, p[i]);
    red[tid] = mx; __syncthreads();
    for (int s = 128; s > 0; s >>= 1) { if (tid < s) red[tid] = fmaxf(red[tid], red[tid + s]); __syncthreads(); }
    mx = red[0]; __syncthreads();
    float sum = 0.0f;
    for (int i = tid; i < n; i += 256) { float e = __expf(p[i] - mx); p[i] = e; sum += e; }
    red[tid] = sum; __syncthreads();
    for (int s = 128; s > 0; s >>= 1) { if (tid < s) red[tid] += red[tid + s]; __syncthreads(); }
    float inv = 1.0f / red[0];
    for (int i = tid; i < n; i += 256) p[i] *= inv;
}

// h = LayerNorm(h + a) in place; one block per row of 256
__global__ void __launch_bounds__(256)
add_layernorm(float* __restrict__ h, const float* __restrict__ a,
              const float* __restrict__ g, const float* __restrict__ b)
{
    __shared__ float red[256];
    long long idx = (long long)blockIdx.x * 256 + threadIdx.x;
    int tid = threadIdx.x;
    float v = h[idx] + a[idx];
    red[tid] = v; __syncthreads();
    for (int s = 128; s > 0; s >>= 1) { if (tid < s) red[tid] += red[tid + s]; __syncthreads(); }
    float mean = red[0] * (1.0f / 256.0f); __syncthreads();
    float d = v - mean;
    red[tid] = d * d; __syncthreads();
    for (int s = 128; s > 0; s >>= 1) { if (tid < s) red[tid] += red[tid + s]; __syncthreads(); }
    float var = red[0] * (1.0f / 256.0f);
    h[idx] = g[tid] * d * rsqrtf(var + 1e-5f) + b[tid];
}

// mean+max pool over L; grid = B*256 blocks; pooled:(B,512)
__global__ void __launch_bounds__(256)
pool_mean_max(const float* __restrict__ ln, float* __restrict__ pooled, int L)
{
    int b = blockIdx.x >> 8;
    int d = blockIdx.x & 255;
    int tid = threadIdx.x;
    __shared__ float rs[256], rm[256];
    float s = 0.0f, m = -3.0e38f;
    for (int l = tid; l < L; l += 256) {
        float v = ln[((long long)b * L + l) * 256 + d];
        s += v; m = fmaxf(m, v);
    }
    rs[tid] = s; rm[tid] = m; __syncthreads();
    for (int st = 128; st > 0; st >>= 1) {
        if (tid < st) { rs[tid] += rs[tid + st]; rm[tid] = fmaxf(rm[tid], rm[tid + st]); }
        __syncthreads();
    }
    if (tid == 0) {
        pooled[b * 512 + d]       = rs[0] / (float)L;
        pooled[b * 512 + 256 + d] = rm[0];
    }
}

// ---------------------------------------------------------------------------
static inline void launch_gemm(const float* A, long long lda, long long sAb, long long sAh,
                               const float* Wt, long long ldw, long long sWb, long long sWh,
                               const float* b1, const float* b2,
                               float* C, long long ldc, long long sCb, long long sCh,
                               int M, int N, int K, float alpha, int act, int Z, int nH,
                               hipStream_t stream)
{
    dim3 grid((N + 63) / 64, (M + 63) / 64, Z);
    gemm_wmma<<<grid, 256, 0, stream>>>(A, lda, sAb, sAh, Wt, ldw, sWb, sWh,
                                        b1, b2, C, ldc, sCb, sCh, M, N, K, alpha, act, nH);
}

static inline void launch_wt(const float* w, float* wt, int N, int K, int dstLd,
                             hipStream_t stream)
{
    transpose_nk<<<(N * K + 255) / 256, 256, 0, stream>>>(w, wt, N, K, dstLd);
}

extern "C" void kernel_launch(void* const* d_in, const int* in_sizes, int n_in,
                              void* d_out, int out_size, void* d_ws, size_t ws_size,
                              hipStream_t stream)
{
    const int B = 2, L = 5500, C0 = 105, C1 = 64, C2 = 32;
    const int M = B * L;                 // 11000
    const long long LL = (long long)L * L;

    // ---- inputs (setup_inputs dict order, nested dicts flattened in order) ----
    int i = 0;
    const float* x       = (const float*)d_in[i++];  // (2,105,5500)
    const float* conv1_w = (const float*)d_in[i++];
    const float* conv1_b = (const float*)d_in[i++];
    const float* bn1_g   = (const float*)d_in[i++];
    const float* bn1_b   = (const float*)d_in[i++];
    const float* conv2_w = (const float*)d_in[i++];
    const float* conv2_b = (const float*)d_in[i++];
    const float* bn2_g   = (const float*)d_in[i++];
    const float* bn2_b   = (const float*)d_in[i++];
    const float* Wih_f0  = (const float*)d_in[i++];
    const float* Whh_f0  = (const float*)d_in[i++];
    const float* bih_f0  = (const float*)d_in[i++];
    const float* bhh_f0  = (const float*)d_in[i++];
    const float* Wih_b0  = (const float*)d_in[i++];
    const float* Whh_b0  = (const float*)d_in[i++];
    const float* bih_b0  = (const float*)d_in[i++];
    const float* bhh_b0  = (const float*)d_in[i++];
    const float* Wih_f1  = (const float*)d_in[i++];
    const float* Whh_f1  = (const float*)d_in[i++];
    const float* bih_f1  = (const float*)d_in[i++];
    const float* bhh_f1  = (const float*)d_in[i++];
    const float* Wih_b1  = (const float*)d_in[i++];
    const float* Whh_b1  = (const float*)d_in[i++];
    const float* bih_b1  = (const float*)d_in[i++];
    const float* bhh_b1  = (const float*)d_in[i++];
    const float* q_w     = (const float*)d_in[i++];
    const float* q_b     = (const float*)d_in[i++];
    const float* k_w     = (const float*)d_in[i++];
    const float* k_b     = (const float*)d_in[i++];
    const float* v_w     = (const float*)d_in[i++];
    const float* v_b     = (const float*)d_in[i++];
    const float* o_w     = (const float*)d_in[i++];
    const float* o_b     = (const float*)d_in[i++];
    const float* ln_g    = (const float*)d_in[i++];
    const float* ln_b    = (const float*)d_in[i++];
    const float* fc1_w   = (const float*)d_in[i++];
    const float* fc1_b   = (const float*)d_in[i++];
    const float* fc2_w   = (const float*)d_in[i++];
    const float* fc2_b   = (const float*)d_in[i++];
    const float* fc3_w   = (const float*)d_in[i++];
    const float* fc3_b   = (const float*)d_in[i++];
    (void)in_sizes; (void)n_in; (void)out_size; (void)ws_size;

    // ---- outputs ----
    float* logits = (float*)d_out;              // (2,95)
    float* attn   = (float*)d_out + 190;        // (2,4,5500,5500)

    // ---- workspace layout (floats), with reuse; ~87 MB total ----
    float* ws   = (float*)d_ws;
    float* h1   = ws;                   //   704000  conv1 out (B,64,L)
    float* h2   = ws + 704000;          //   352000  conv2 out (B,32,L)
    float* xl   = ws + 1056000;         //   352000  (B,L,32)
    float* preF = ws + 1408000;         //  5632000  (B,L,512)
    float* preB = ws + 7040000;         //  5632000
    float* out0 = ws + 12672000;        //  2816000  (B,L,256)
    float* out1 = ws + 15488000;        //  2816000
    // transposed (K-major) weights
    float* wtp      = ws + 18304000;
    float* Wih_f0T  = wtp;              //  16384  (32x512)
    float* Wih_b0T  = wtp + 16384;      //  16384
    float* Wih_f1T  = wtp + 32768;      // 131072  (256x512)
    float* Wih_b1T  = wtp + 163840;     // 131072
    float* q_wT     = wtp + 294912;     //  65536  (256x256)
    float* k_wT     = wtp + 360448;     //  65536
    float* v_wT     = wtp + 425984;     //  65536
    float* o_wT     = wtp + 491520;     //  65536
    float* fc1T     = wtp + 557056;     // 131072  (512x256)
    float* fc2T     = wtp + 688128;     //  32768  (256x128)
    float* fc3T     = wtp + 720896;     //  16384  (128x128, ld padded to 128)
    float* KtB      = ws + 19041280;    // 2816000 (B,256,L) = K^T per batch
    // reuse after recurrences / conv stages:
    float* Qb      = preF;              //  2816000
    float* Kb      = preF + 2816000;    //  2816000
    float* Vb      = preB;              //  2816000
    float* ctx     = preB + 2816000;    //  2816000
    float* attnout = out0;              //  2816000 (out0 free after layer-1 pre GEMMs)
    float* pooled  = h1;                //  1024
    float* z1      = h1 + 1024;         //  512
    float* z2      = h1 + 1536;         //  256

    // 0) one-time weight transposes to K-major (so GEMM B-loads are contiguous)
    launch_wt(Wih_f0, Wih_f0T, 512,  32, 512, stream);
    launch_wt(Wih_b0, Wih_b0T, 512,  32, 512, stream);
    launch_wt(Wih_f1, Wih_f1T, 512, 256, 512, stream);
    launch_wt(Wih_b1, Wih_b1T, 512, 256, 512, stream);
    launch_wt(q_w,   q_wT, 256, 256, 256, stream);
    launch_wt(k_w,   k_wT, 256, 256, 256, stream);
    launch_wt(v_w,   v_wT, 256, 256, 256, stream);
    launch_wt(o_w,   o_wT, 256, 256, 256, stream);
    launch_wt(fc1_w, fc1T, 256, 512, 256, stream);
    launch_wt(fc2_w, fc2T, 128, 256, 128, stream);
    launch_wt(fc3_w, fc3T,  95, 128, 128, stream);   // ld padded to 128

    // 1-2) convs + BN + ReLU
    conv_bn_relu<<<(B * C1 * L + 255) / 256, 256, 0, stream>>>(
        x, conv1_w, conv1_b, bn1_g, bn1_b, h1, B, C0, C1, L);
    conv_bn_relu<<<(B * C2 * L + 255) / 256, 256, 0, stream>>>(
        h1, conv2_w, conv2_b, bn2_g, bn2_b, h2, B, C1, C2, L);
    // 3) (B,32,L) -> (B,L,32)
    nchw_to_nlc<<<(B * C2 * L + 255) / 256, 256, 0, stream>>>(h2, xl, B, C2, L);

    // 4-5) LSTM layer 0 input projections (WMMA)
    launch_gemm(xl, 32, 0, 0, Wih_f0T, 512, 0, 0, bih_f0, bhh_f0,
                preF, 512, 0, 0, M, 512, 32, 1.0f, 0, 1, 1, stream);
    launch_gemm(xl, 32, 0, 0, Wih_b0T, 512, 0, 0, bih_b0, bhh_b0,
                preB, 512, 0, 0, M, 512, 32, 1.0f, 0, 1, 1, stream);
    // 6) layer 0 recurrence (fwd + bwd blocks run concurrently)
    lstm_rec<<<2, 256, 0, stream>>>(preF, preB, Whh_f0, Whh_b0, out0, L);

    // 7-8) LSTM layer 1 input projections
    launch_gemm(out0, 256, 0, 0, Wih_f1T, 512, 0, 0, bih_f1, bhh_f1,
                preF, 512, 0, 0, M, 512, 256, 1.0f, 0, 1, 1, stream);
    launch_gemm(out0, 256, 0, 0, Wih_b1T, 512, 0, 0, bih_b1, bhh_b1,
                preB, 512, 0, 0, M, 512, 256, 1.0f, 0, 1, 1, stream);
    // 9) layer 1 recurrence
    lstm_rec<<<2, 256, 0, stream>>>(preF, preB, Whh_f1, Whh_b1, out1, L);

    // 10-12) Q,K,V projections
    launch_gemm(out1, 256, 0, 0, q_wT, 256, 0, 0, q_b, nullptr,
                Qb, 256, 0, 0, M, 256, 256, 1.0f, 0, 1, 1, stream);
    launch_gemm(out1, 256, 0, 0, k_wT, 256, 0, 0, k_b, nullptr,
                Kb, 256, 0, 0, M, 256, 256, 1.0f, 0, 1, 1, stream);
    launch_gemm(out1, 256, 0, 0, v_wT, 256, 0, 0, v_b, nullptr,
                Vb, 256, 0, 0, M, 256, 256, 1.0f, 0, 1, 1, stream);

    // 12b) K -> K^T per batch: (B,L,256) -> (B,256,L)
    transpose_ld<<<(B * L * 256 + 255) / 256, 256, 0, stream>>>(Kb, KtB, B, L, 256);

    // 13) scores = Q K^T / 8 -> straight into d_out attn region (z = b*4+h)
    launch_gemm(Qb, 256, (long long)L * 256, 64,
                KtB, L, 256LL * L, 64LL * L,
                nullptr, nullptr,
                attn, L, 4 * LL, LL,
                L, L, 64, 0.125f, 0, 8, 4, stream);

    // 14) softmax rows in place (B*H*L rows)
    softmax_rows<<<B * 4 * L, 256, 0, stream>>>(attn, L);

    // 15) ctx = attn @ V  (V is naturally K-major: element (k,n) = V[k*256 + h*64 + n])
    launch_gemm(attn, L, 4 * LL, LL,
                Vb, 256, (long long)L * 256, 64,
                nullptr, nullptr,
                ctx, 256, (long long)L * 256, 64,
                L, 64, L, 1.0f, 0, 8, 4, stream);

    // 16) output projection
    launch_gemm(ctx, 256, 0, 0, o_wT, 256, 0, 0, o_b, nullptr,
                attnout, 256, 0, 0, M, 256, 256, 1.0f, 0, 1, 1, stream);

    // 17) residual + LayerNorm in place on out1
    add_layernorm<<<M, 256, 0, stream>>>(out1, attnout, ln_g, ln_b);

    // 18) mean+max pool -> (2,512)
    pool_mean_max<<<B * 256, 256, 0, stream>>>(out1, pooled, L);

    // 19-21) classifier head
    launch_gemm(pooled, 512, 0, 0, fc1T, 256, 0, 0, fc1_b, nullptr,
                z1, 256, 0, 0, B, 256, 512, 1.0f, 1, 1, 1, stream);
    launch_gemm(z1, 256, 0, 0, fc2T, 128, 0, 0, fc2_b, nullptr,
                z2, 128, 0, 0, B, 128, 256, 1.0f, 1, 1, 1, stream);
    launch_gemm(z2, 128, 0, 0, fc3T, 128, 0, 0, fc3_b, nullptr,
                logits, 95, 0, 0, B, 95, 128, 1.0f, 0, 1, 1, stream);
}
